// AST_GAT_86595130622201
// MI455X (gfx1250) — matrix operand order repaired and checked by reading, not compiled, not run on hardware
//
#include <hip/hip_runtime.h>
#include <hip/hip_bf16.h>
#include <cstdint>

// ---------------------------------------------------------------------------
// MI455X (gfx1250) implementation notes:
//  * All dense math runs through v_wmma_f32_16x16x32_bf16 (fp32 accumulate).
//  * Weights are pre-swizzled once into the B-operand lane/VGPR layout,
//    stored half-split ([kt][half][lane][4dw]) so LDS reads are b128 and
//    bank-conflict-free (lane stride = 4 dwords).
//  * GEMM blocks are 128 threads / 4 waves / 64 rows. The B tile for the
//    current column block lives in LDS, double-buffered, shared by all 4
//    waves => 4x less L2 weight traffic than per-wave streaming. Next-next
//    tile is prefetched with __builtin_prefetch (global_prefetch_b8).
//  * A-operands (fp32 in memory) are staged per block into LDS as bf16 with
//    pitch 392 (784B rows -> bank stride 4), read in contiguous k order (k = 16h + s).
//  * Segment sums use float atomics (in-degree ~1 for the combine graph).
//  * The reference's scan `done` latch lives in a device int; every loop
//    kernel reads it, matching jax.lax.scan gating deterministically.
// ---------------------------------------------------------------------------

#define DIM 384
#define KT  12          // 384 / 32  K-tiles
#define JT  24          // 384 / 16  N-tiles
#define ROWS_PB 64      // rows per GEMM block (4 waves x 16 rows)
#define LDS_PITCH 392   // bf16 elements per LDS row (padding kills conflicts)
#define GEMM_THREADS 128
#define WSZ_MAT 73728   // 24*12*32*8 dwords per swizzled 384x384 bf16 matrix
#define BT_U4 768       // uint4 per (jt, term) B tile: 12kt * 2half * 32lane
#define N_ITERS 20

typedef __attribute__((ext_vector_type(16))) __bf16 v16bf;
typedef __attribute__((ext_vector_type(8)))  float  v8f;

struct __align__(16) U4 { uint32_t x, y, z, w; };
struct __align__(16) B32 { U4 lo, hi; };

static __device__ __forceinline__ v16bf pack16(U4 lo, U4 hi) {
  B32 t{lo, hi};
  return __builtin_bit_cast(v16bf, t);
}

static __device__ __forceinline__ uint32_t f2bf_bits(float f) {
  union { float f; uint32_t u; } v; v.f = f;
  uint32_t u = v.u;
  return (u + 0x7FFFu + ((u >> 16) & 1u)) >> 16;   // RNE truncation to bf16
}
static __device__ __forceinline__ uint16_t f2bf(float f) {
  return (uint16_t)f2bf_bits(f);
}

// --- weight pre-swizzle: W[k][n] fp32 (optionally Wa+Wb) -> WMMA-B bf16x2 ---
// dword order: [jt][kt][half][lane][vv 0..3], element v = half*4 + vv
__global__ void k_build_wswz(const float* __restrict__ Wa,
                             const float* __restrict__ Wb,
                             uint32_t* __restrict__ dst) {
  int t = blockIdx.x * blockDim.x + threadIdx.x;
  if (t >= JT * KT * 32 * 8) return;
  int vv   = t & 3;
  int lane = (t >> 2) & 31;
  int hh   = (t >> 7) & 1;
  int kt   = (t >> 8) % KT;
  int jt   = (t >> 8) / KT;
  int v  = hh * 4 + vv;
  int n  = jt * 16 + (lane & 15);
  int k0 = kt * 32 + (lane >> 4) * 16 + 2 * v;
  float e0 = Wa[(size_t)k0 * DIM + n];
  float e1 = Wa[(size_t)(k0 + 1) * DIM + n];
  if (Wb) { e0 += Wb[(size_t)k0 * DIM + n]; e1 += Wb[(size_t)(k0 + 1) * DIM + n]; }
  dst[t] = f2bf_bits(e0) | (f2bf_bits(e1) << 16);
}

__global__ void k_bias_add(const float* a, const float* b, float* o) {
  int t = blockIdx.x * blockDim.x + threadIdx.x;
  if (t < DIM) o[t] = a[t] + b[t];
}

// --- fused GEMM: Y[R,384] (=/+=) A0@W0 (+ A1@W1) (+ bias) ------------------
__global__ __launch_bounds__(GEMM_THREADS)
void k_gemm(const float* __restrict__ A0, const uint32_t* __restrict__ W0,
            const float* __restrict__ A1, const uint32_t* __restrict__ W1,
            const float* __restrict__ bias, float* __restrict__ Y,
            int R, int accumulate, const int* __restrict__ doneFlag) {
  if (doneFlag && doneFlag[0]) return;       // uniform early-out (scan gate)
  extern __shared__ __align__(16) uint16_t smem[];
  const int row0 = blockIdx.x * ROWS_PB;
  if (row0 >= R) return;

  uint16_t* sA0 = smem;                                  // 64 x 392 bf16
  uint16_t* sA1 = smem + ROWS_PB * LDS_PITCH;            // 64 x 392 bf16
  uint32_t* sB  = (uint32_t*)(sA1 + ROWS_PB * LDS_PITCH); // 2 buf x 2 term x 3072 dw

  for (int idx = threadIdx.x; idx < ROWS_PB * DIM; idx += GEMM_THREADS) {
    int r = idx / DIM, c = idx - r * DIM;
    int rr = row0 + r; if (rr >= R) rr = R - 1;
    sA0[r * LDS_PITCH + c] = f2bf(A0[(size_t)rr * DIM + c]);
    if (A1) sA1[r * LDS_PITCH + c] = f2bf(A1[(size_t)rr * DIM + c]);
  }

  auto stageB = [&](int jt, int buf) {
    const U4* g0 = (const U4*)W0 + (size_t)jt * BT_U4;
    U4* d0 = (U4*)(sB + (size_t)buf * 2 * 4 * BT_U4);
    for (int i = threadIdx.x; i < BT_U4; i += GEMM_THREADS) d0[i] = g0[i];
    if (A1) {
      const U4* g1 = (const U4*)W1 + (size_t)jt * BT_U4;
      U4* d1 = d0 + BT_U4;
      for (int i = threadIdx.x; i < BT_U4; i += GEMM_THREADS) d1[i] = g1[i];
    }
  };

  stageB(0, 0);
  __syncthreads();

  const int lane = threadIdx.x & 31;
  const int wave = threadIdx.x >> 5;
  const int h    = lane >> 4;        // K-half select (contiguous k = 16h + s for A and B)
  const int n16  = lane & 15;        // A row / C column within tile
  const U4* a0row = (const U4*)(sA0 + (size_t)(wave * 16 + n16) * LDS_PITCH);
  const U4* a1row = (const U4*)(sA1 + (size_t)(wave * 16 + n16) * LDS_PITCH);

  for (int jt = 0; jt < JT; ++jt) {
    const int cb = jt & 1;
    if (jt + 1 < JT) stageB(jt + 1, cb ^ 1);
    if (jt + 2 < JT) {       // prefetch jt+2 B tiles (global_prefetch_b8)
      const char* p0 = (const char*)((const U4*)W0 + (size_t)(jt + 2) * BT_U4);
      __builtin_prefetch(p0 + (size_t)threadIdx.x * 128, 0, 1);
      if (A1) {
        const char* p1 = (const char*)((const U4*)W1 + (size_t)(jt + 2) * BT_U4);
        __builtin_prefetch(p1 + (size_t)threadIdx.x * 128, 0, 1);
      }
    }

    const U4* bb = (const U4*)(sB + (size_t)cb * 2 * 4 * BT_U4);
    v8f acc = {};
#pragma unroll
    for (int kt = 0; kt < KT; ++kt) {
      v16bf a = pack16(a0row[kt * 4 + 2 * h], a0row[kt * 4 + 2 * h + 1]);
      v16bf b = pack16(bb[kt * 64 + lane], bb[kt * 64 + 32 + lane]);
      acc = __builtin_amdgcn_wmma_f32_16x16x32_bf16(false, a, false, b,
                                                    (short)0, acc, false, false);
    }
    if (A1) {
      const U4* bb1 = bb + BT_U4;
#pragma unroll
      for (int kt = 0; kt < KT; ++kt) {
        v16bf a = pack16(a1row[kt * 4 + 2 * h], a1row[kt * 4 + 2 * h + 1]);
        v16bf b = pack16(bb1[kt * 64 + lane], bb1[kt * 64 + 32 + lane]);
        acc = __builtin_amdgcn_wmma_f32_16x16x32_bf16(false, a, false, b,
                                                      (short)0, acc, false, false);
      }
    }
    const int col = jt * 16 + n16;
    const float bv = bias ? bias[col] : 0.0f;
#pragma unroll
    for (int r = 0; r < 8; ++r) {          // C layout: lanes<16 -> M=r, else M=8+r
      int m = (h ? 8 : 0) + r;
      int row = row0 + wave * 16 + m;
      if (row < R) {
        size_t o = (size_t)row * DIM + col;
        float val = acc[r] + bv;
        if (accumulate) Y[o] += val; else Y[o] = val;
      }
    }
    __syncthreads();    // next iter reads the buffer staged this iter
  }
}

// --- small helpers ----------------------------------------------------------
__global__ void k_zero_f(float* p, size_t n) {
  size_t i = (size_t)blockIdx.x * blockDim.x + threadIdx.x;
  if (i < n) p[i] = 0.0f;
}
__global__ void k_zero_u(uint32_t* p, size_t n) {
  size_t i = (size_t)blockIdx.x * blockDim.x + threadIdx.x;
  if (i < n) p[i] = 0u;
}
__global__ void k_gather_subx(const float* __restrict__ code_emb,
                              const int* __restrict__ ids,
                              float* __restrict__ subx, int n_sub) {
  size_t t = (size_t)blockIdx.x * blockDim.x + threadIdx.x;
  if (t >= (size_t)n_sub * DIM) return;
  int i = (int)(t / DIM), c = (int)(t - (size_t)i * DIM);
  subx[t] = code_emb[(size_t)ids[i] * DIM + c];
}
// phase-B: both hetero segment sums in one pass over all edges
__global__ void k_scatter_init(const float* __restrict__ varx,
                               const float* __restrict__ subx,
                               const int* __restrict__ E, int M, int n_sub,
                               float* __restrict__ s_comp,
                               float* __restrict__ s_comb) {
  int t = blockIdx.x * blockDim.x + threadIdx.x;
  if (t >= M * 96) return;
  int e = t / 96, c = (t - e * 96) * 4;
  int src = E[e], dst = E[M + e];
  const float* sv; float* dv;
  if (src >= n_sub) { sv = varx + (size_t)(src - n_sub) * DIM; dv = s_comp; }
  else              { sv = subx + (size_t)src * DIM;           dv = s_comb; }
  float4 v = *(const float4*)(sv + c);
  float* o = dv + (size_t)dst * DIM + c;
  atomicAdd(o + 0, v.x); atomicAdd(o + 1, v.y);
  atomicAdd(o + 2, v.z); atomicAdd(o + 3, v.w);
}
__global__ void k_init_active(const int* __restrict__ E, int* act, int M,
                              int n_sub, int* done, int* anyCnt) {
  int e = blockIdx.x * blockDim.x + threadIdx.x;
  if (e < M) act[e] = (E[e] < n_sub) ? 1 : 0;
  if (blockIdx.x == 0 && threadIdx.x == 0) { *done = 0; *anyCnt = 0; }
}
__global__ void k_loop_scatter(const float* __restrict__ x,
                               const int* __restrict__ E,
                               const int* __restrict__ act, int M, int n_sub,
                               float* __restrict__ ssum, float* __restrict__ cnt,
                               const int* __restrict__ done) {
  if (done[0]) return;
  int t = blockIdx.x * blockDim.x + threadIdx.x;
  if (t >= M * 96) return;
  int e = t / 96, c = (t - e * 96) * 4;
  if (!act[e]) return;
  int src = E[e], dst = E[M + e];           // active => src internal
  float4 v = *(const float4*)(x + (size_t)src * DIM + c);
  float* o = ssum + (size_t)dst * DIM + c;
  atomicAdd(o + 0, v.x); atomicAdd(o + 1, v.y);
  atomicAdd(o + 2, v.z); atomicAdd(o + 3, v.w);
  if (c == 0) atomicAdd(cnt + dst, 1.0f);
}
__global__ void k_mean(float* __restrict__ ssum, const float* __restrict__ cnt,
                       size_t n, const int* __restrict__ done) {
  if (done[0]) return;
  size_t t = (size_t)blockIdx.x * blockDim.x + threadIdx.x;
  if (t >= n) return;
  ssum[t] /= fmaxf(cnt[t / DIM], 1.0f);
}
__global__ void k_commit(const float* __restrict__ xn, float* __restrict__ x,
                         size_t n4, const int* __restrict__ done) {
  if (done[0]) return;
  size_t t = (size_t)blockIdx.x * blockDim.x + threadIdx.x;
  if (t >= n4) return;
  ((float4*)x)[t] = ((const float4*)xn)[t];
}
__global__ void k_mark(const int* __restrict__ E, const int* __restrict__ act,
                       int M, uint32_t* __restrict__ is_tgt,
                       const int* __restrict__ done) {
  if (done[0]) return;
  int e = blockIdx.x * blockDim.x + threadIdx.x;
  if (e >= M) return;
  if (act[e]) is_tgt[E[M + e]] = 1u;
}
__global__ void k_prune(const int* __restrict__ E, const int* __restrict__ act,
                        int* __restrict__ actN, int M, int n_sub,
                        const uint32_t* __restrict__ is_tgt, int* anyCnt,
                        const int* __restrict__ done) {
  int e = blockIdx.x * blockDim.x + threadIdx.x;
  if (e >= M) return;
  int a = act[e];
  if (done[0]) { actN[e] = a; return; }     // frozen once done
  int an = 0;
  if (a) {
    int src = E[e];
    int sl = (src >= n_sub) ? (src - n_sub) : src;   // src_loc
    if (sl > n_sub - 1) sl = n_sub - 1;              // clip
    an = is_tgt[sl] ? 1 : 0;
  }
  actN[e] = an;
  if (an) atomicAdd(anyCnt, 1);
}
__global__ void k_done_update(int* done, int* anyCnt) {
  if (anyCnt[0] == 0) done[0] = 1;
  anyCnt[0] = 0;
}
__global__ void k_gather_out(const float* __restrict__ x,
                             const int* __restrict__ roots,
                             float* __restrict__ out, int B) {
  int t = blockIdx.x * blockDim.x + threadIdx.x;
  if (t >= B * DIM) return;
  int b = t / DIM, c = t - b * DIM;
  out[t] = x[(size_t)roots[b] * DIM + c];
}

// ---------------------------------------------------------------------------
extern "C" void kernel_launch(void* const* d_in, const int* in_sizes, int n_in,
                              void* d_out, int out_size, void* d_ws, size_t ws_size,
                              hipStream_t stream) {
  const float* var_feats = (const float*)d_in[0];
  const float* code_emb  = (const float*)d_in[1];
  const float* W_lin = (const float*)d_in[2];
  const float* b_lin = (const float*)d_in[3];
  const float* Wl_c  = (const float*)d_in[4];
  const float* bl_c  = (const float*)d_in[5];
  const float* Wr_c  = (const float*)d_in[6];
  const float* Wl_b  = (const float*)d_in[7];
  const float* bl_b  = (const float*)d_in[8];
  const float* Wr_b  = (const float*)d_in[9];
  const float* Wl_lp = (const float*)d_in[10];
  const float* bl_lp = (const float*)d_in[11];
  const float* Wr_lp = (const float*)d_in[12];
  const int* subcode_ids = (const int*)d_in[13];
  const int* E        = (const int*)d_in[14];
  const int* root_ids = (const int*)d_in[15];

  const int n_var = in_sizes[0] / DIM;     // 65536
  const int n_sub = in_sizes[13];          // 65536
  const int M     = in_sizes[14] / 2;      // 131008
  const int Bq    = in_sizes[15];          // 64

  const size_t NF = (size_t)n_sub * DIM;
  const size_t VF = (size_t)n_var * DIM;
  const size_t big = (VF > NF) ? VF : NF;

  // workspace carve-up (all 16B aligned)
  char* w = (char*)d_ws;
  float* buf0 = (float*)w; w += big * 4;           // var_x -> x
  float* buf1 = (float*)w; w += big * 4;           // sub_x -> x_next
  float* buf2 = (float*)w; w += big * 4;           // s_comp -> ssum/mean
  float* buf3 = (float*)w; w += big * 4;           // s_comb
  float* cnt  = (float*)w; w += (size_t)n_sub * 4;
  uint32_t* Wswz = (uint32_t*)w; w += 6ull * WSZ_MAT * 4;
  float* bias_c = (float*)w; w += DIM * 4;
  int* actA = (int*)w; w += (size_t)M * 4;
  int* actB = (int*)w; w += (size_t)M * 4;
  uint32_t* is_tgt = (uint32_t*)w; w += (size_t)n_sub * 4;
  int* anyCnt = (int*)w; w += 16;
  int* done   = (int*)w; w += 16;
  (void)ws_size; (void)n_in;

  const int T = 256;
  const int wgrid  = (JT * KT * 32 * 8 + T - 1) / T;
  const int egrid  = (M + T - 1) / T;
  const int ecgrid = (M * 96 + T - 1) / T;
  const int nfgrid = (int)((NF + T - 1) / T);

  // 0. weight swizzles (bf16 B-operand layout) + combined bias
  k_build_wswz<<<wgrid, T, 0, stream>>>(W_lin, nullptr, Wswz + 0 * WSZ_MAT);
  k_build_wswz<<<wgrid, T, 0, stream>>>(Wl_c,  nullptr, Wswz + 1 * WSZ_MAT);
  k_build_wswz<<<wgrid, T, 0, stream>>>(Wr_c,  Wr_b,    Wswz + 2 * WSZ_MAT);
  k_build_wswz<<<wgrid, T, 0, stream>>>(Wl_b,  nullptr, Wswz + 3 * WSZ_MAT);
  k_build_wswz<<<wgrid, T, 0, stream>>>(Wl_lp, nullptr, Wswz + 4 * WSZ_MAT);
  k_build_wswz<<<wgrid, T, 0, stream>>>(Wr_lp, nullptr, Wswz + 5 * WSZ_MAT);
  k_bias_add<<<2, 192, 0, stream>>>(bl_c, bl_b, bias_c);

  // LDS: A (2 x 64 x 392 bf16) + B double-buffer (2 x 2 x 3072 dwords)
  const size_t lds = 2ull * ROWS_PB * LDS_PITCH * 2 + 2ull * 2 * 4 * BT_U4 * 4;
  auto gemm = [&](const float* A0, const uint32_t* W0, const float* A1,
                  const uint32_t* W1, const float* bias, float* Y, int R,
                  int acc, const int* df) {
    k_gemm<<<dim3((R + ROWS_PB - 1) / ROWS_PB), dim3(GEMM_THREADS), lds, stream>>>(
        A0, W0, A1, W1, bias, Y, R, acc, df);
  };

  // 1. node features
  k_gather_subx<<<nfgrid, T, 0, stream>>>(code_emb, subcode_ids, buf1, n_sub);
  gemm(var_feats, Wswz + 0 * WSZ_MAT, nullptr, nullptr, b_lin, buf0, n_var, 0, nullptr);

  // 2. hetero segment sums
  k_zero_f<<<nfgrid, T, 0, stream>>>(buf2, NF);
  k_zero_f<<<nfgrid, T, 0, stream>>>(buf3, NF);
  k_scatter_init<<<ecgrid, T, 0, stream>>>(buf0, buf1, E, M, n_sub, buf2, buf3);

  // 3. x = s_comp@Wl_c + sub_x@(Wr_c+Wr_b) + (bl_c+bl_b)  (+= s_comb@Wl_b)
  gemm(buf2, Wswz + 1 * WSZ_MAT, buf1, Wswz + 2 * WSZ_MAT, bias_c, buf0, n_sub, 0, nullptr);
  gemm(buf3, Wswz + 3 * WSZ_MAT, nullptr, nullptr, nullptr, buf0, n_sub, 1, nullptr);

  // 4. 20 gated mean-aggregate SAGE iterations
  k_init_active<<<egrid, T, 0, stream>>>(E, actA, M, n_sub, done, anyCnt);
  int* cur = actA; int* nxt = actB;
  for (int it = 0; it < N_ITERS; ++it) {
    k_zero_f<<<nfgrid, T, 0, stream>>>(buf2, NF);
    k_zero_f<<<(n_sub + T - 1) / T, T, 0, stream>>>(cnt, (size_t)n_sub);
    k_loop_scatter<<<ecgrid, T, 0, stream>>>(buf0, E, cur, M, n_sub, buf2, cnt, done);
    k_mean<<<nfgrid, T, 0, stream>>>(buf2, cnt, NF, done);
    gemm(buf2, Wswz + 4 * WSZ_MAT, buf0, Wswz + 5 * WSZ_MAT, bl_lp, buf1, n_sub, 0, done);
    k_commit<<<(int)((NF / 4 + T - 1) / T), T, 0, stream>>>(buf1, buf0, NF / 4, done);
    k_zero_u<<<(n_sub + T - 1) / T, T, 0, stream>>>(is_tgt, (size_t)n_sub);
    k_mark<<<egrid, T, 0, stream>>>(E, cur, M, is_tgt, done);
    k_prune<<<egrid, T, 0, stream>>>(E, cur, nxt, M, n_sub, is_tgt, anyCnt, done);
    k_done_update<<<1, 1, 0, stream>>>(done, anyCnt);
    int* tswap = cur; cur = nxt; nxt = tswap;
  }

  // 5. gather roots
  k_gather_out<<<(Bq * DIM + T - 1) / T, T, 0, stream>>>(buf0, root_ids, (float*)d_out, Bq);
}
